// FeatureMamba_2525440770252
// MI455X (gfx1250) — matrix operand order repaired and checked by reading, not compiled, not run on hardware
//
#include <hip/hip_runtime.h>
#include <hip/hip_bf16.h>
#include <math.h>

#define BB      16
#define EE      2048
#define MM      256
#define NLAYER  4
#define DSTATE  16
#define DCONV   4
#define DINNER  512
#define DTRANK  16
#define ROWS    (BB*EE)          /* 32768 */

typedef __attribute__((ext_vector_type(16))) __bf16         v16bf;
typedef __attribute__((ext_vector_type(8)))  float          v8f;
typedef __attribute__((ext_vector_type(16))) unsigned short v16u;
typedef __attribute__((ext_vector_type(8)))  unsigned short v8u;

static __device__ __forceinline__ unsigned short f2bf(float f) {
  union { float f; unsigned int u; } c; c.f = f;
  unsigned int u = c.u;
  u += 0x7FFFu + ((u >> 16) & 1u);   // round-to-nearest-even
  return (unsigned short)(u >> 16);
}
static __device__ __forceinline__ float sigmoidf_(float x) { return 1.0f / (1.0f + __expf(-x)); }
static __device__ __forceinline__ float siluf_(float x)    { return x * sigmoidf_(x); }

// Cubic B-spline bases, uniform knots g(i)=0.4*i-2.2 (GRID=5, ORDER=3) -> 8 bases
static __device__ __forceinline__ void bspline8(float x, float* outb) {
  float b[11];
#pragma unroll
  for (int i = 0; i < 11; ++i) {
    float gi  = 0.4f * i - 2.2f;
    float gi1 = 0.4f * (i + 1) - 2.2f;
    b[i] = (x >= gi && x < gi1) ? 1.0f : 0.0f;
  }
#pragma unroll
  for (int k = 1; k <= 3; ++k) {
    float inv = 1.0f / (0.4f * (float)k);
#pragma unroll
    for (int i = 0; i + k < 11; ++i) {
      float gi   = 0.4f * i - 2.2f;
      float gik1 = 0.4f * (i + k + 1) - 2.2f;
      b[i] = (x - gi) * inv * b[i] + (gik1 - x) * inv * b[i + 1];
    }
  }
#pragma unroll
  for (int g = 0; g < 8; ++g) outb[g] = b[g];
}

// ---------------------------------------------------------------------------
// C(MrxN) = A(MrxK, bf16 row-major) @ W(NxK, bf16 row-major)^T, f32 accumulate.
// One wave = one 16x16 tile; K stepped by 32 through v_wmma_f32_16x16x32_bf16.
// ---------------------------------------------------------------------------
__global__ void gemm_bf16_wmma(const unsigned short* __restrict__ A,
                               const unsigned short* __restrict__ W,
                               float* __restrict__ C,
                               int Mr, int N, int K) {
  int wave = (int)((blockIdx.x * blockDim.x + threadIdx.x) >> 5);
  int lane = threadIdx.x & 31;
  int Nt = N >> 4;
  int total = (Mr >> 4) * Nt;
  if (wave >= total) return;
  int tm   = wave / Nt;
  int tn   = wave - tm * Nt;
  int half = lane >> 4;
  int lr   = lane & 15;

  const unsigned short* arow = A + (size_t)(tm * 16 + lr) * (size_t)K;
  const unsigned short* wrow = W + (size_t)(tn * 16 + lr) * (size_t)K;

  v8f acc = {0.f, 0.f, 0.f, 0.f, 0.f, 0.f, 0.f, 0.f};
  for (int k0 = 0; k0 < K; k0 += 32) {
    // A fragment: element e -> k = (e>>3)*16 + half*8 + (e&7)  (two 16B runs)
    v8u a0 = *(const v8u*)(arow + k0 + half * 8);
    v8u a1 = *(const v8u*)(arow + k0 + 16 + half * 8);
    // B fragment: element e -> k = (e>>3)*16 + half*8 + (e&7), column n = lr (two 16B runs of W row)
    v8u b0 = *(const v8u*)(wrow + k0 + half * 8);
    v8u b1 = *(const v8u*)(wrow + k0 + 16 + half * 8);
    v16u au = __builtin_shufflevector(a0, a1, 0,1,2,3,4,5,6,7,8,9,10,11,12,13,14,15);
    v16u bu = __builtin_shufflevector(b0, b1, 0,1,2,3,4,5,6,7,8,9,10,11,12,13,14,15);
    acc = __builtin_amdgcn_wmma_f32_16x16x32_bf16(
        false, __builtin_bit_cast(v16bf, au),
        false, __builtin_bit_cast(v16bf, bu),
        (short)0, acc, false, false);
  }
  // D layout: VGPR r -> M = tm*16 + half*8 + r, N = tn*16 + lr
  float* cbase = C + (size_t)(tm * 16 + half * 8) * (size_t)N + tn * 16 + lr;
#pragma unroll
  for (int r = 0; r < 8; ++r) cbase[(size_t)r * N] = acc[r];
}

__global__ void cvt_bf16_kernel(const float* __restrict__ src,
                                unsigned short* __restrict__ dst, int n) {
  int i = blockIdx.x * blockDim.x + threadIdx.x;
  if (i < n) dst[i] = f2bf(src[i]);
}

// KAN 1->M expand: t[row, o] = silu(x)*bw[o] + (sum_g basis_g(x) * sw[o,g]) * sc[o]
__global__ void kan1_kernel(const float* __restrict__ x,
                            const float* __restrict__ bw,
                            const float* __restrict__ sw,
                            const float* __restrict__ sc,
                            float* __restrict__ t,
                            unsigned short* __restrict__ tbf) {
  int row = blockIdx.x;           // b*EE + e
  int o   = threadIdx.x;          // 0..MM-1
  float xv = x[row];
  float bas[8];
  bspline8(xv, bas);
  float sp = 0.f;
#pragma unroll
  for (int g = 0; g < 8; ++g) sp += bas[g] * sw[o * 8 + g];
  float v = siluf_(xv) * bw[o] + sp * sc[o];
  size_t idx = (size_t)row * MM + o;
  t[idx]   = v;
  tbf[idx] = f2bf(v);
}

// depthwise causal conv (DC=4) over L on the xc half of xz, then SiLU
__global__ void conv_silu_kernel(const float* __restrict__ xz,
                                 const float* __restrict__ cw,
                                 const float* __restrict__ cb,
                                 float* __restrict__ u,
                                 unsigned short* __restrict__ ubf) {
  long i = (long)blockIdx.x * blockDim.x + threadIdx.x;  // over ROWS*DINNER
  if (i >= (long)ROWS * DINNER) return;
  int  d    = (int)(i & (DINNER - 1));
  long row  = i >> 9;
  int  ll   = (int)(row & (EE - 1));
  long brow = row - ll;                                  // b*EE
  float acc = cb[d];
#pragma unroll
  for (int j = 0; j < DCONV; ++j) {
    int lsrc = ll - (DCONV - 1) + j;
    if (lsrc >= 0) acc += cw[d * DCONV + j] * xz[(brow + lsrc) * (2 * DINNER) + d];
  }
  float v = siluf_(acc);
  u[i]   = v;
  ubf[i] = f2bf(v);
}

// delta = softplus(dt @ dtw^T + dtb), dt = xdbl[:, :16]
__global__ void dt_softplus_kernel(const float* __restrict__ xdbl,
                                   const float* __restrict__ dtw,
                                   const float* __restrict__ dtb,
                                   float* __restrict__ delta) {
  long i = (long)blockIdx.x * blockDim.x + threadIdx.x;
  if (i >= (long)ROWS * DINNER) return;
  int  d   = (int)(i & (DINNER - 1));
  long row = i >> 9;
  const float* xr = xdbl + row * 48;
  float acc = dtb[d];
#pragma unroll
  for (int r = 0; r < DTRANK; ++r) acc += xr[r] * dtw[d * DTRANK + r];
  delta[i] = (acc > 20.f) ? acc : log1pf(__expf(acc));
}

// Sequential selective scan: one thread per (b,d); h[16], A-row in VGPRs.
// y is written into the (dead) xc columns of xz.
__global__ void scan_kernel(const float* __restrict__ delta,
                            const float* __restrict__ u,
                            const float* __restrict__ xdbl,
                            const float* __restrict__ A_log,
                            float* __restrict__ ydst /* = xz */) {
  int tid = blockIdx.x * blockDim.x + threadIdx.x;
  if (tid >= BB * DINNER) return;
  int d = tid & (DINNER - 1);
  int b = tid >> 9;
  float a[DSTATE], h[DSTATE];
#pragma unroll
  for (int s = 0; s < DSTATE; ++s) {
    a[s] = -__expf(A_log[d * DSTATE + s]);
    h[s] = 0.f;
  }
  long base = (long)b * EE;
  for (int l = 0; l < EE; ++l) {
    long row = base + l;
    float dtv = delta[row * DINNER + d];
    float ut  = u[row * DINNER + d];
    float du  = dtv * ut;
    const float* xr = xdbl + row * 48;   // B at +16, C at +32 (wave-uniform)
    float yv = 0.f;
#pragma unroll
    for (int s = 0; s < DSTATE; ++s) {
      float e = __expf(dtv * a[s]);
      h[s] = e * h[s] + du * xr[DTRANK + s];
      yv  += h[s] * xr[DTRANK + DSTATE + s];
    }
    ydst[row * (2 * DINNER) + d] = yv;
  }
}

// y = (y + u*D) * silu(z) -> bf16 for out_proj
__global__ void gate_kernel(const float* __restrict__ xz,
                            const float* __restrict__ u,
                            const float* __restrict__ Dp,
                            unsigned short* __restrict__ ybf) {
  long i = (long)blockIdx.x * blockDim.x + threadIdx.x;
  if (i >= (long)ROWS * DINNER) return;
  int  d   = (int)(i & (DINNER - 1));
  long row = i >> 9;
  float yv = xz[row * (2 * DINNER) + d];
  float z  = xz[row * (2 * DINNER) + DINNER + d];
  float v  = (yv + u[i] * Dp[d]) * siluf_(z);
  ybf[i] = f2bf(v);
}

// t = rmsnorm(t2 + t, nw), also produce bf16 copy
__global__ void rmsnorm_kernel(const float* __restrict__ t2,
                               const float* __restrict__ tin,
                               const float* __restrict__ nw,
                               float* __restrict__ tout,
                               unsigned short* __restrict__ tbf) {
  __shared__ float sdata[256];
  long row = blockIdx.x;
  int  o   = threadIdx.x;
  size_t idx = (size_t)row * MM + o;
  float v = t2[idx] + tin[idx];
  sdata[o] = v * v;
  __syncthreads();
  for (int s2 = 128; s2 > 0; s2 >>= 1) {
    if (o < s2) sdata[o] += sdata[o + s2];
    __syncthreads();
  }
  float rs = rsqrtf(sdata[0] / (float)MM + 1.1920929e-07f);
  float w  = v * rs * nw[o];
  tout[idx] = w;
  tbf[idx]  = f2bf(w);
}

// KAN M->1 contract: s[row] = sum_o silu(t)*bw[o] + (sum_g basis*sw[o,g])*sc[o]
__global__ void kan2_kernel(const float* __restrict__ t,
                            const float* __restrict__ bw,
                            const float* __restrict__ sw,
                            const float* __restrict__ sc,
                            float* __restrict__ sout) {
  __shared__ float sdata[256];
  long row = blockIdx.x;
  int  o   = threadIdx.x;
  float xv = t[(size_t)row * MM + o];
  float bas[8];
  bspline8(xv, bas);
  float sp = 0.f;
#pragma unroll
  for (int g = 0; g < 8; ++g) sp += bas[g] * sw[o * 8 + g];
  float contrib = siluf_(xv) * bw[o] + sp * sc[o];
  sdata[o] = contrib;
  __syncthreads();
  for (int s2 = 128; s2 > 0; s2 >>= 1) {
    if (o < s2) sdata[o] += sdata[o + s2];
    __syncthreads();
  }
  if (o == 0) sout[row] = sdata[0];
}

// out = rmsnorm(s.reshape(B,1,E) + x, nxw) over E per batch row
__global__ void final_kernel(const float* __restrict__ s,
                             const float* __restrict__ x,
                             const float* __restrict__ nxw,
                             float* __restrict__ out) {
  __shared__ float sdata[256];
  int b   = blockIdx.x;
  int tid = threadIdx.x;
  float v[8];
  float ss = 0.f;
#pragma unroll
  for (int j = 0; j < 8; ++j) {
    int e = tid + j * 256;
    v[j] = s[(size_t)b * EE + e] + x[(size_t)b * EE + e];
    ss += v[j] * v[j];
  }
  sdata[tid] = ss;
  __syncthreads();
  for (int s2 = 128; s2 > 0; s2 >>= 1) {
    if (tid < s2) sdata[tid] += sdata[tid + s2];
    __syncthreads();
  }
  float rs = rsqrtf(sdata[0] / (float)EE + 1.1920929e-07f);
#pragma unroll
  for (int j = 0; j < 8; ++j) {
    int e = tid + j * 256;
    out[(size_t)b * EE + e] = v[j] * rs * nxw[e];
  }
}

extern "C" void kernel_launch(void* const* d_in, const int* in_sizes, int n_in,
                              void* d_out, int out_size, void* d_ws, size_t ws_size,
                              hipStream_t stream) {
  const float* x     = (const float*)d_in[0];
  const float* k1bw  = (const float*)d_in[1];
  const float* k1sw  = (const float*)d_in[2];
  const float* k1sc  = (const float*)d_in[3];
  const float* k2bw  = (const float*)d_in[4];
  const float* k2sw  = (const float*)d_in[5];
  const float* k2sc  = (const float*)d_in[6];
  const float* ipw   = (const float*)d_in[7];
  const float* cw    = (const float*)d_in[8];
  const float* cb    = (const float*)d_in[9];
  const float* xpw   = (const float*)d_in[10];
  const float* dtw   = (const float*)d_in[11];
  const float* dtb   = (const float*)d_in[12];
  const float* A_log = (const float*)d_in[13];
  const float* Dp    = (const float*)d_in[14];
  const float* opw   = (const float*)d_in[15];
  const float* nw    = (const float*)d_in[16];
  const float* nxw   = (const float*)d_in[17];
  float* out = (float*)d_out;

  char*  ws  = (char*)d_ws;
  size_t off = 0;
  auto alloc = [&](size_t bytes) -> char* {
    char* p = ws + off;
    off = (off + bytes + 255) & ~(size_t)255;
    return p;
  };
  float*          t     = (float*)          alloc((size_t)ROWS * MM * 4);
  float*          t2    = (float*)          alloc((size_t)ROWS * MM * 4);
  unsigned short* tbf   = (unsigned short*) alloc((size_t)ROWS * MM * 2);
  float*          xz    = (float*)          alloc((size_t)ROWS * 1024 * 4);
  float*          u     = (float*)          alloc((size_t)ROWS * DINNER * 4);
  unsigned short* ubf   = (unsigned short*) alloc((size_t)ROWS * DINNER * 2);
  float*          xdbl  = (float*)          alloc((size_t)ROWS * 48 * 4);
  float*          delta = (float*)          alloc((size_t)ROWS * DINNER * 4);
  unsigned short* ybf   = (unsigned short*) alloc((size_t)ROWS * DINNER * 2);
  float*          svec  = (float*)          alloc((size_t)ROWS * 4);
  unsigned short* ipwb  = (unsigned short*) alloc((size_t)NLAYER * 1024 * MM * 2);
  unsigned short* xpwb  = (unsigned short*) alloc((size_t)NLAYER * 48 * DINNER * 2);
  unsigned short* opwb  = (unsigned short*) alloc((size_t)NLAYER * MM * DINNER * 2);

  // Weight f32 -> bf16 conversion (recomputed every call; deterministic)
  {
    int n;
    n = NLAYER * 1024 * MM;
    cvt_bf16_kernel<<<(n + 255) / 256, 256, 0, stream>>>(ipw, ipwb, n);
    n = NLAYER * 48 * DINNER;
    cvt_bf16_kernel<<<(n + 255) / 256, 256, 0, stream>>>(xpw, xpwb, n);
    n = NLAYER * MM * DINNER;
    cvt_bf16_kernel<<<(n + 255) / 256, 256, 0, stream>>>(opw, opwb, n);
  }

  kan1_kernel<<<ROWS, 256, 0, stream>>>(x, k1bw, k1sw, k1sc, t, tbf);

  long n1 = (long)ROWS * DINNER;
  for (int l = 0; l < NLAYER; ++l) {
    {
      int tiles = (ROWS / 16) * (1024 / 16);
      gemm_bf16_wmma<<<(tiles * 32 + 127) / 128, 128, 0, stream>>>(
          tbf, ipwb + (size_t)l * 1024 * MM, xz, ROWS, 1024, MM);
    }
    conv_silu_kernel<<<(int)((n1 + 255) / 256), 256, 0, stream>>>(
        xz, cw + (size_t)l * DINNER * DCONV, cb + (size_t)l * DINNER, u, ubf);
    {
      int tiles = (ROWS / 16) * (48 / 16);
      gemm_bf16_wmma<<<(tiles * 32 + 127) / 128, 128, 0, stream>>>(
          ubf, xpwb + (size_t)l * 48 * DINNER, xdbl, ROWS, 48, DINNER);
    }
    dt_softplus_kernel<<<(int)((n1 + 255) / 256), 256, 0, stream>>>(
        xdbl, dtw + (size_t)l * DINNER * DTRANK, dtb + (size_t)l * DINNER, delta);
    scan_kernel<<<(BB * DINNER + 127) / 128, 128, 0, stream>>>(
        delta, u, xdbl, A_log + (size_t)l * DINNER * DSTATE, xz);
    gate_kernel<<<(int)((n1 + 255) / 256), 256, 0, stream>>>(
        xz, u, Dp + (size_t)l * DINNER, ybf);
    {
      int tiles = (ROWS / 16) * (MM / 16);
      gemm_bf16_wmma<<<(tiles * 32 + 127) / 128, 128, 0, stream>>>(
          ybf, opwb + (size_t)l * MM * DINNER, t2, ROWS, MM, DINNER);
    }
    rmsnorm_kernel<<<ROWS, 256, 0, stream>>>(t2, t, nw, t, tbf);
  }

  kan2_kernel<<<ROWS, 256, 0, stream>>>(t, k2bw, k2sw, k2sc, svec);
  final_kernel<<<BB, 256, 0, stream>>>(svec, x, nxw, out);

  (void)in_sizes; (void)n_in; (void)out_size; (void)ws_size;
}
